// FStaticLoss_28166395527342
// MI455X (gfx1250) — matrix operand order repaired and checked by reading, not compiled, not run on hardware
//
#include <hip/hip_runtime.h>

typedef __attribute__((ext_vector_type(16))) _Float16 v16h;
typedef __attribute__((ext_vector_type(4)))  _Float16 v4h;
typedef __attribute__((ext_vector_type(8)))  float    v8f;

constexpr int D          = 256;   // feature dim
constexpr int KCLS       = 64;    // classes
constexpr int TILE       = 32;    // rows per tile (WMMA K dim)
constexpr int LDS_STRIDE = D + 16;              // halfs, padded vs bank conflicts
constexpr int WSTRIDE    = KCLS * D + 2 * KCLS; // floats per per-block partial slice

// registers staging one 32x256 tile slice for this thread (software pipeline)
struct TileRegs {
    float4 f[8];   // 32 contiguous floats of one row
    int    c;      // class id (threads 0..31 only)
};

__device__ __forceinline__ void load_tile(const float* __restrict__ z,
                                          const long long* __restrict__ cls,
                                          int t, int row, int q, int tid,
                                          TileRegs& r)
{
    const float4* src = (const float4*)(z + ((size_t)t * TILE + row) * D + q * 32);
#pragma unroll
    for (int u = 0; u < 8; ++u) r.f[u] = src[u];
    if (tid < TILE) r.c = (int)cls[(size_t)t * TILE + tid];
}

__device__ __forceinline__ void store_tile(const TileRegs& r, int row, int q, int tid,
                                           _Float16 (*zh)[LDS_STRIDE],
                                           float (*pnorm)[8], int* cls_s)
{
    float nrm = 0.0f;
#pragma unroll
    for (int u = 0; u < 8; ++u) {
        float4 f = r.f[u];
        nrm += f.x * f.x + f.y * f.y + f.z * f.z + f.w * f.w;
        v4h hv;
        hv[0] = (_Float16)f.x; hv[1] = (_Float16)f.y;
        hv[2] = (_Float16)f.z; hv[3] = (_Float16)f.w;
        *(v4h*)&zh[row][q * 32 + u * 4] = hv;   // ds_store_b64
    }
    pnorm[row][q] = nrm;
    if (tid < TILE) cls_s[tid] = r.c;
}

// ---------------------------------------------------------------------------
// Pass 1: per-block partial segment sums via one-hot WMMA, plus counts/sumsq.
// sums[K,D] = OneHot[K,Ntile] x Z[Ntile,D] with v_wmma_f32_16x16x32_f16
// (A = one-hot, exact in f16; B = z in f16; C accumulates in f32).
// Software-pipelined: tile t+G global loads issue during tile t's WMMA phase.
// Fully deterministic: no atomics; fixed-order reductions only.
// ---------------------------------------------------------------------------
__global__ __launch_bounds__(256) void seg_accum(const float* __restrict__ z,
                                                 const long long* __restrict__ cls,
                                                 float* __restrict__ ws,
                                                 int tiles)
{
    __shared__ _Float16 zh[TILE][LDS_STRIDE];  // f16 tile, row-major
    __shared__ float    pnorm[TILE][8];        // per-row partial |z|^2
    __shared__ int      cls_s[TILE];

    const int tid  = threadIdx.x;
    const int lane = tid & 31;
    const int wv   = tid >> 5;      // wave 0..7 -> owns cols [32*wv, 32*wv+32)
    const int h    = lane >> 4;     // lane half (WMMA layout)
    const int ln   = lane & 15;
    const int row  = tid >> 3;      // 0..31 (8 threads per row for staging)
    const int q    = tid & 7;       // 32-col chunk within the row

    const int G = (int)gridDim.x;

    // persistent f32 accumulators: 4 class-groups x 2 col-tiles x 8 VGPRs
    v8f acc[4][2];
#pragma unroll
    for (int g = 0; g < 4; ++g)
#pragma unroll
        for (int ct = 0; ct < 2; ++ct)
#pragma unroll
            for (int e = 0; e < 8; ++e)
                acc[g][ct][e] = 0.0f;

    float my_cnt = 0.0f, my_ssq = 0.0f;   // class `tid` (tid < 64)

    // ---- prologue: stage first tile ---------------------------------------
    int t0 = blockIdx.x;
    if (t0 < tiles) {
        TileRegs cur;
        load_tile(z, cls, t0, row, q, tid, cur);
        store_tile(cur, row, q, tid, zh, pnorm, cls_s);
    }
    __syncthreads();

    for (int t = t0; t < tiles; t += G) {
        const bool has_next = (t + G) < tiles;   // block-uniform
        TileRegs nxt;
        if (has_next) load_tile(z, cls, t + G, row, q, tid, nxt);  // in flight

        // ---- counts / sumsq: thread k owns class k, fixed scan order ------
        if (tid < KCLS) {
            for (int r = 0; r < TILE; ++r) {
                if (cls_s[r] == tid) {
                    my_cnt += 1.0f;
                    float s = 0.0f;
#pragma unroll
                    for (int u = 0; u < 8; ++u) s += pnorm[r][u];
                    my_ssq += s;
                }
            }
        }

        // ---- WMMA phase (uniform control flow, EXEC all ones) -------------
        // B (32x16 f16): element i of v16h holds K = 16*h + i, col = base+ln
        v16h b0, b1;
#pragma unroll
        for (int i = 0; i < 16; ++i) {
            b0[i] = zh[16 * h + i][wv * 32 + ln];
            b1[i] = zh[16 * h + i][wv * 32 + 16 + ln];
        }
        // A (16x32 one-hot f16): element i holds K = 16*h + i.
        // Build all four class-group A matrices first (independent registers,
        // lets the scheduler fill WMMA hazard slots with real work).
        v16h amat[4];
#pragma unroll
        for (int g = 0; g < 4; ++g) {
#pragma unroll
            for (int i = 0; i < 16; ++i) {
                int kk = 16 * h + i;
                amat[g][i] = (cls_s[kk] == g * 16 + ln) ? (_Float16)1.0f
                                                        : (_Float16)0.0f;
            }
        }
#pragma unroll
        for (int g = 0; g < 4; ++g) {
            acc[g][0] = __builtin_amdgcn_wmma_f32_16x16x32_f16(
                false, amat[g], false, b0, (short)0, acc[g][0], false, false);
            acc[g][1] = __builtin_amdgcn_wmma_f32_16x16x32_f16(
                false, amat[g], false, b1, (short)0, acc[g][1], false, false);
        }

        __syncthreads();   // all waves done reading zh/pnorm/cls_s
        if (has_next) store_tile(nxt, row, q, tid, zh, pnorm, cls_s);
        __syncthreads();   // next tile visible to all waves
    }

    // ---- flush per-block partials (plain stores, no atomics) --------------
    float* wsb = ws + (size_t)blockIdx.x * WSTRIDE;
#pragma unroll
    for (int g = 0; g < 4; ++g)
#pragma unroll
        for (int ct = 0; ct < 2; ++ct)
#pragma unroll
            for (int r = 0; r < 8; ++r) {
                int m   = g * 16 + 8 * h + r;            // class id
                int col = wv * 32 + ct * 16 + ln;        // feature col
                wsb[m * D + col] = acc[g][ct][r];
            }
    if (tid < KCLS) {
        wsb[KCLS * D + tid]        = my_cnt;
        wsb[KCLS * D + KCLS + tid] = my_ssq;
    }
}

// ---------------------------------------------------------------------------
// Pass 2: reduce G partial slices, finalize centroids and variances.
// var_k = ssq_k / max(cnt,1) - ||centroid_k||^2   (exact algebraic fold)
// ---------------------------------------------------------------------------
__global__ __launch_bounds__(256) void seg_finalize(const float* __restrict__ ws,
                                                    float* __restrict__ out,
                                                    int G)
{
    __shared__ float red[256];
    __shared__ float cnt_s, ssq_s;

    const int k = blockIdx.x;   // class
    const int d = threadIdx.x;  // feature

    float s = 0.0f;
    for (int g = 0; g < G; ++g)
        s += ws[(size_t)g * WSTRIDE + k * D + d];

    if (d == 0) {
        float cn = 0.0f, sq = 0.0f;
        for (int g = 0; g < G; ++g) {
            cn += ws[(size_t)g * WSTRIDE + KCLS * D + k];
            sq += ws[(size_t)g * WSTRIDE + KCLS * D + KCLS + k];
        }
        cnt_s = cn; ssq_s = sq;
    }
    __syncthreads();

    const float safe = fmaxf(cnt_s, 1.0f);
    const float cen  = s / safe;
    out[k * D + d] = cen;                 // centroid

    red[d] = cen * cen;
    __syncthreads();
#pragma unroll
    for (int off = 128; off > 0; off >>= 1) {
        if (d < off) red[d] += red[d + off];
        __syncthreads();
    }
    if (d == 0) out[KCLS * D + k] = ssq_s / safe - red[0];  // variance
}

// ---------------------------------------------------------------------------
extern "C" void kernel_launch(void* const* d_in, const int* in_sizes, int n_in,
                              void* d_out, int out_size, void* d_ws, size_t ws_size,
                              hipStream_t stream)
{
    const float*     z   = (const float*)d_in[0];
    const long long* c   = (const long long*)d_in[1];
    float*           out = (float*)d_out;
    float*           ws  = (float*)d_ws;

    const int N     = in_sizes[1];       // 131072 rows
    const int tiles = N / TILE;          // 4096 tiles of 32 rows

    // one partial slice per block; cap by available workspace
    int G = (int)(ws_size / (sizeof(float) * (size_t)WSTRIDE));
    if (G > 256) G = 256;
    if (G < 1)   G = 1;

    seg_accum<<<dim3(G), dim3(256), 0, stream>>>(z, c, ws, tiles);
    seg_finalize<<<dim3(KCLS), dim3(256), 0, stream>>>(ws, out, G);
}
